// TransitionModel_26817775796445
// MI455X (gfx1250) — matrix operand order repaired and checked by reading, not compiled, not run on hardware
//
#include <hip/hip_runtime.h>
#include <hip/hip_bf16.h>

// ---------------------------------------------------------------------------
// RSSM (Dreamer transition model) for MI455X / gfx1250, wave32 + WMMA bf16.
// All GEMMs via v_wmma_f32_16x16x32_bf16. Weights packed once per launch into
// WMMA B-fragment order (lane-contiguous 32B). Activations kept row-major
// bf16 so A fragments load as two 16B chunks per lane. K-loop is software-
// pipelined with pointer-increment addressing and unconditional speculative
// prefetch (OOB prefetches are dropped by HW) so the loop body is branch-free
// and VMEM overlaps the matrix pipe.
// ---------------------------------------------------------------------------

typedef __attribute__((ext_vector_type(16))) __bf16 v16bf;
typedef __attribute__((ext_vector_type(8)))  __bf16 v8bf;
typedef __attribute__((ext_vector_type(8)))  float  v8f;

#define TT  50
#define BB  2048
#define SS  128
#define BEL 512
#define HH  1024
#define EE  1024
#define AACT 6
#define KEMB 160          // S+A=134 padded to multiple of 32
#define LN_EPS 0.001f
#define UPDATE_BIAS (-1.0f)

enum { ACT_NONE = 0, ACT_RELU = 1, ACT_ELU = 2 };

// ---------------------------------------------------------------------------
// Weight pack: fp32 [K,N] row-major -> bf16 B-fragments.
// Fragment for (ktile, ntile): lane l holds 16 contiguous bf16:
//   k = kt*32 + (l>=16 ? 8 : 0) + e + (e>=8 ? 8 : 0), n = nt*16 + (l&15), e = 0..15.
// ---------------------------------------------------------------------------
__global__ void pack_weights(const float* __restrict__ W, __bf16* __restrict__ out,
                             int K, int Kpad, int N) {
    int idx   = blockIdx.x * blockDim.x + threadIdx.x;
    int total = (Kpad >> 5) * (N >> 4) * 32;
    if (idx >= total) return;
    int lane   = idx & 31;
    int tile   = idx >> 5;
    int ntiles = N >> 4;
    int nt     = tile % ntiles;
    int kt     = tile / ntiles;
    int n      = nt * 16 + (lane & 15);
    int kbase  = kt * 32 + ((lane >> 4) << 3);
    __bf16* dst = out + ((size_t)tile * 32 + lane) * 16;
#pragma unroll
    for (int e = 0; e < 16; ++e) {
        int k = kbase + e + (e >= 8 ? 8 : 0);
        float v = (k < K) ? W[(size_t)k * N + n] : 0.0f;
        dst[e] = (__bf16)v;
    }
}

// ---------------------------------------------------------------------------
// WMMA GEMM: C[M=2048, N] = act(A[M,Kpad] @ Wpacked + bias)
// block = 256 threads = 8 waves; wave tile = 32(M) x 64(N) -> block 256 x 64.
// Software-pipelined K loop: tile kt+1 loaded while tile kt's 8 WMMAs issue.
// ---------------------------------------------------------------------------
__device__ __forceinline__ v16bf load_afrag(const __bf16* __restrict__ ap) {
    v8bf alo = *(const v8bf*)(ap);
    v8bf ahi = *(const v8bf*)(ap + 16);
    v16bf a;
#pragma unroll
    for (int i = 0; i < 8; ++i) { a[i] = alo[i]; a[i + 8] = ahi[i]; }
    return a;
}

__global__ __launch_bounds__(256)
void gemm_wmma_bf16(const __bf16* __restrict__ A, int lda,
                    const __bf16* __restrict__ Bp,
                    const float*  __restrict__ bias,
                    int Kpad, int N, int act,
                    float*  __restrict__ outF,
                    __bf16* __restrict__ outB,
                    int ostride) {
    const int lane  = threadIdx.x & 31;
    const int wave  = threadIdx.x >> 5;
    const int m0    = blockIdx.y * 256 + wave * 32;
    const int n0    = blockIdx.x * 64;
    const int ntile = n0 >> 4;
    const int ntiles = N >> 4;

    const int arow0 = m0 + (lane & 15);        // M-tile 0
    const int arow1 = arow0 + 16;              // M-tile 1
    const int kbase = (lane >> 4) << 3;        // 0 or 8 (ISA 16-bit A layout)

    v8f acc[8];
#pragma unroll
    for (int j = 0; j < 8; ++j) acc[j] = (v8f){0.f,0.f,0.f,0.f,0.f,0.f,0.f,0.f};

    const int ktiles = Kpad >> 5;
    const size_t bstep = (size_t)ntiles * 512;   // elems per k-tile of packed B

    // Incremented pointers: A advances 32 elems (64B), B one k-tile per iter.
    const __bf16* aptr0 = A + (size_t)arow0 * lda + kbase;
    const __bf16* aptr1 = A + (size_t)arow1 * lda + kbase;
    const __bf16* bptr  = Bp + ((size_t)ntile * 32 + lane) * 16;

    // ---- prologue: load k-tile 0 ----
    v16bf a0 = load_afrag(aptr0);
    v16bf a1 = load_afrag(aptr1);
    v16bf b0 = *(const v16bf*)(bptr);
    v16bf b1 = *(const v16bf*)(bptr + 512);
    v16bf b2 = *(const v16bf*)(bptr + 1024);
    v16bf b3 = *(const v16bf*)(bptr + 1536);

    for (int kt = 0; kt < ktiles - 1; ++kt) {
        aptr0 += 32; aptr1 += 32; bptr += bstep;
        // Unconditional speculative prefetch of k-tile kt+2 (HW drops OOB).
        __builtin_prefetch(bptr + bstep, 0, 1);
        // load next k-tile while current WMMAs execute
        v16bf na0 = load_afrag(aptr0);
        v16bf na1 = load_afrag(aptr1);
        v16bf nb0 = *(const v16bf*)(bptr);
        v16bf nb1 = *(const v16bf*)(bptr + 512);
        v16bf nb2 = *(const v16bf*)(bptr + 1024);
        v16bf nb3 = *(const v16bf*)(bptr + 1536);

        acc[0] = __builtin_amdgcn_wmma_f32_16x16x32_bf16(false, a0, false, b0, (short)0, acc[0], false, false);
        acc[1] = __builtin_amdgcn_wmma_f32_16x16x32_bf16(false, a0, false, b1, (short)0, acc[1], false, false);
        acc[2] = __builtin_amdgcn_wmma_f32_16x16x32_bf16(false, a0, false, b2, (short)0, acc[2], false, false);
        acc[3] = __builtin_amdgcn_wmma_f32_16x16x32_bf16(false, a0, false, b3, (short)0, acc[3], false, false);
        acc[4] = __builtin_amdgcn_wmma_f32_16x16x32_bf16(false, a1, false, b0, (short)0, acc[4], false, false);
        acc[5] = __builtin_amdgcn_wmma_f32_16x16x32_bf16(false, a1, false, b1, (short)0, acc[5], false, false);
        acc[6] = __builtin_amdgcn_wmma_f32_16x16x32_bf16(false, a1, false, b2, (short)0, acc[6], false, false);
        acc[7] = __builtin_amdgcn_wmma_f32_16x16x32_bf16(false, a1, false, b3, (short)0, acc[7], false, false);

        a0 = na0; a1 = na1; b0 = nb0; b1 = nb1; b2 = nb2; b3 = nb3;
    }
    // ---- epilogue k-tile ----
    acc[0] = __builtin_amdgcn_wmma_f32_16x16x32_bf16(false, a0, false, b0, (short)0, acc[0], false, false);
    acc[1] = __builtin_amdgcn_wmma_f32_16x16x32_bf16(false, a0, false, b1, (short)0, acc[1], false, false);
    acc[2] = __builtin_amdgcn_wmma_f32_16x16x32_bf16(false, a0, false, b2, (short)0, acc[2], false, false);
    acc[3] = __builtin_amdgcn_wmma_f32_16x16x32_bf16(false, a0, false, b3, (short)0, acc[3], false, false);
    acc[4] = __builtin_amdgcn_wmma_f32_16x16x32_bf16(false, a1, false, b0, (short)0, acc[4], false, false);
    acc[5] = __builtin_amdgcn_wmma_f32_16x16x32_bf16(false, a1, false, b1, (short)0, acc[5], false, false);
    acc[6] = __builtin_amdgcn_wmma_f32_16x16x32_bf16(false, a1, false, b2, (short)0, acc[6], false, false);
    acc[7] = __builtin_amdgcn_wmma_f32_16x16x32_bf16(false, a1, false, b3, (short)0, acc[7], false, false);

    // Epilogue: C layout -> lane<16: N=lane, M=base+i ; lane>=16: N=lane-16, M=base+8+i
    const int colb = lane & 15;
    const int rofs = (lane >> 4) << 3;
#pragma unroll
    for (int j = 0; j < 4; ++j) {
        int n = n0 + j * 16 + colb;
        float bv = bias[n];
#pragma unroll
        for (int mt = 0; mt < 2; ++mt) {
#pragma unroll
            for (int i = 0; i < 8; ++i) {
                int r = m0 + mt * 16 + rofs + i;
                float v = acc[mt * 4 + j][i] + bv;
                if (act == ACT_RELU)      v = v > 0.f ? v : 0.f;
                else if (act == ACT_ELU)  v = v > 0.f ? v : expm1f(v);
                if (outF) outF[(size_t)r * ostride + n] = v;
                if (outB) outB[(size_t)r * ostride + n] = (__bf16)v;
            }
        }
    }
}

// ---------------------------------------------------------------------------
// Build embed input A0[B, 160] = bf16([state*nt, action, 0-pad])
// ---------------------------------------------------------------------------
__global__ void build_a0(const float* __restrict__ state,
                         const float* __restrict__ act_t,
                         const float* __restrict__ nt_t,
                         __bf16* __restrict__ A0) {
    int idx = blockIdx.x * blockDim.x + threadIdx.x;
    if (idx >= BB * KEMB) return;
    int b = idx / KEMB, c = idx % KEMB;
    float v;
    if (c < SS)             v = state[b * SS + c] * nt_t[b];
    else if (c < SS + AACT) v = act_t[b * AACT + (c - SS)];
    else                    v = 0.f;
    A0[idx] = (__bf16)v;
}

// fp32 [rows,cols] -> bf16 into dst[r*dstride + dcol + c]
__global__ void cvt_f32_bf16(const float* __restrict__ src, __bf16* __restrict__ dst,
                             int rows, int cols, int dstride, int dcol) {
    int idx = blockIdx.x * blockDim.x + threadIdx.x;
    if (idx >= rows * cols) return;
    int r = idx / cols, c = idx % cols;
    dst[(size_t)r * dstride + dcol + c] = (__bf16)src[idx];
}

// ---------------------------------------------------------------------------
// Fused LayerNorm(1536) + GRU gates + belief update. One block per batch row.
// ---------------------------------------------------------------------------
__global__ __launch_bounds__(256)
void gru_ln_kernel(const float* __restrict__ parts,
                   const float* __restrict__ ln_g, const float* __restrict__ ln_b,
                   float* __restrict__ belief,        // carry [B,512] (in/out)
                   float* __restrict__ out_belief,    // d_out slice for step t
                   __bf16* __restrict__ A2,           // p1 input [B,512]
                   __bf16* __restrict__ A3) {         // q1 input [B,1536], cols 0..511
    __shared__ float s1[256], s2[256];
    int r = blockIdx.x;
    int tid = threadIdx.x;
    const float* x = parts + (size_t)r * 1536;

    float a = 0.f, b = 0.f;
#pragma unroll
    for (int i = 0; i < 6; ++i) {
        float v = x[tid + 256 * i];
        a += v; b += v * v;
    }
    s1[tid] = a; s2[tid] = b;
    __syncthreads();
    for (int off = 128; off > 0; off >>= 1) {
        if (tid < off) { s1[tid] += s1[tid + off]; s2[tid] += s2[tid + off]; }
        __syncthreads();
    }
    float mean = s1[0] * (1.f / 1536.f);
    float var  = s2[0] * (1.f / 1536.f) - mean * mean;
    float inv  = rsqrtf(var + LN_EPS);

#pragma unroll
    for (int p = 0; p < 2; ++p) {
        int j = tid + 256 * p;                  // 0..511
        float rv = (x[j]        - mean) * inv * ln_g[j]        + ln_b[j];
        float cv = (x[j + 512]  - mean) * inv * ln_g[j + 512]  + ln_b[j + 512];
        float uv = (x[j + 1024] - mean) * inv * ln_g[j + 1024] + ln_b[j + 1024];
        float reset  = 1.f / (1.f + expf(-rv));
        float cand   = tanhf(reset * cv);
        float update = 1.f / (1.f + expf(-(uv + UPDATE_BIAS)));
        float bold = belief[(size_t)r * BEL + j];
        float bn = update * cand + (1.f - update) * bold;
        belief[(size_t)r * BEL + j]     = bn;
        out_belief[(size_t)r * BEL + j] = bn;
        A2[(size_t)r * BEL + j]  = (__bf16)bn;
        A3[(size_t)r * 1536 + j] = (__bf16)bn;
    }
}

// ---------------------------------------------------------------------------
// Softmax(128) + threefry2x32 Gumbel argmax -> probs + one-hot sample.
// One wave (32 lanes x 4 cols) per batch row.
// ---------------------------------------------------------------------------
__device__ __forceinline__ unsigned rotl32(unsigned x, int n) {
    return (x << n) | (x >> (32 - n));
}
__device__ __forceinline__ void threefry2x32(unsigned k0, unsigned k1,
                                             unsigned c0, unsigned c1,
                                             unsigned& o0, unsigned& o1) {
    const unsigned R[8] = {13, 15, 26, 6, 17, 29, 16, 24};
    unsigned ks2 = 0x1BD11BDAu ^ k0 ^ k1;
    unsigned x0 = c0 + k0, x1 = c1 + k1;
#pragma unroll
    for (int i = 0; i < 20; ++i) {
        x0 += x1; x1 = rotl32(x1, R[i & 7]); x1 ^= x0;
        int rr = i + 1;
        if ((rr & 3) == 0) {
            int s = rr >> 2;                     // 1..5
            unsigned ka = (s % 3 == 1) ? k1 : ((s % 3 == 2) ? ks2 : k0);
            unsigned kb = (s % 3 == 1) ? ks2 : ((s % 3 == 2) ? k0 : k1);
            x0 += ka; x1 += kb + (unsigned)s;
        }
    }
    o0 = x0; o1 = x1;
}

__global__ __launch_bounds__(256)
void softmax_sample(const float* __restrict__ logits,
                    const unsigned* __restrict__ key2,
                    float* __restrict__ out_probs,
                    float* __restrict__ out_s,
                    float* __restrict__ carry_state) {
    int lane = threadIdx.x & 31;
    int wave = threadIdx.x >> 5;
    int row  = blockIdx.x * 8 + wave;
    if (row >= BB) return;
    const float* lrow = logits + (size_t)row * 128;
    unsigned k0 = key2[0], k1 = key2[1];

    float v[4];
    float mx = -3.4e38f;
#pragma unroll
    for (int i = 0; i < 4; ++i) {
        v[i] = lrow[lane + 32 * i];
        mx = fmaxf(mx, v[i]);
    }
#pragma unroll
    for (int off = 16; off > 0; off >>= 1) mx = fmaxf(mx, __shfl_xor(mx, off, 32));

    float e[4], s = 0.f;
#pragma unroll
    for (int i = 0; i < 4; ++i) { e[i] = expf(v[i] - mx); s += e[i]; }
#pragma unroll
    for (int off = 16; off > 0; off >>= 1) s += __shfl_xor(s, off, 32);
    float invs = 1.f / s;

    float bs = -3.4e38f; int bi = 0;
#pragma unroll
    for (int i = 0; i < 4; ++i) {
        int c = lane + 32 * i;
        out_probs[(size_t)row * 128 + c] = e[i] * invs;
        unsigned r0, r1;
        threefry2x32(k0, k1, (unsigned)row, (unsigned)c, r0, r1);
        float u = __uint_as_float(0x3F800000u | (r0 >> 9)) - 1.0f;  // [0,1)
        float inner = fmaxf(-logf(fmaxf(u, 1e-12f)), 1e-20f);
        float g = -logf(inner);
        float score = v[i] + g;
        if (score > bs) { bs = score; bi = c; }
    }
#pragma unroll
    for (int off = 16; off > 0; off >>= 1) {
        float os = __shfl_xor(bs, off, 32);
        int   oi = __shfl_xor(bi, off, 32);
        if (os > bs || (os == bs && oi < bi)) { bs = os; bi = oi; }
    }
#pragma unroll
    for (int i = 0; i < 4; ++i) {
        int c = lane + 32 * i;
        float oh = (c == bi) ? 1.f : 0.f;
        out_s[(size_t)row * 128 + c] = oh;
        if (carry_state) carry_state[(size_t)row * 128 + c] = oh;
    }
}

// ---------------------------------------------------------------------------
// Host side
// ---------------------------------------------------------------------------
static inline dim3 gemm_grid(int N) { return dim3(N / 64, BB / 256); }

extern "C" void kernel_launch(void* const* d_in, const int* in_sizes, int n_in,
                              void* d_out, int out_size, void* d_ws, size_t ws_size,
                              hipStream_t stream) {
    (void)in_sizes; (void)n_in; (void)out_size; (void)ws_size;
    const float* prev_state   = (const float*)d_in[0];
    const float* actions      = (const float*)d_in[1];
    const float* prev_belief  = (const float*)d_in[2];
    const float* observations = (const float*)d_in[3];
    const float* nonterm      = (const float*)d_in[4];
    const float* W_embed = (const float*)d_in[5];
    const float* b_embed = (const float*)d_in[6];
    const float* W_gru   = (const float*)d_in[7];
    const float* b_gru   = (const float*)d_in[8];
    const float* ln_g    = (const float*)d_in[9];
    const float* ln_b    = (const float*)d_in[10];
    const float* W_p1    = (const float*)d_in[11];
    const float* b_p1    = (const float*)d_in[12];
    const float* W_p2    = (const float*)d_in[13];
    const float* b_p2    = (const float*)d_in[14];
    const float* W_q1    = (const float*)d_in[15];
    const float* b_q1    = (const float*)d_in[16];
    const float* W_q2    = (const float*)d_in[17];
    const float* b_q2    = (const float*)d_in[18];
    const unsigned* keys_prior = (const unsigned*)d_in[19];
    const unsigned* keys_post  = (const unsigned*)d_in[20];

    char* ws = (char*)d_ws;
    size_t off = 0;
    auto alloc = [&](size_t bytes) -> void* {
        void* p = ws + off;
        off = (off + bytes + 255) & ~(size_t)255;
        return p;
    };
    __bf16* pWe  = (__bf16*)alloc((size_t)KEMB * BEL * 2);
    __bf16* pWg  = (__bf16*)alloc((size_t)1024 * 1536 * 2);
    __bf16* pWp1 = (__bf16*)alloc((size_t)512 * 1024 * 2);
    __bf16* pWp2 = (__bf16*)alloc((size_t)1024 * 128 * 2);
    __bf16* pWq1 = (__bf16*)alloc((size_t)1536 * 1024 * 2);
    __bf16* pWq2 = (__bf16*)alloc((size_t)1024 * 128 * 2);
    __bf16* A0 = (__bf16*)alloc((size_t)BB * KEMB * 2);
    __bf16* A1 = (__bf16*)alloc((size_t)BB * 1024 * 2);
    __bf16* A2 = (__bf16*)alloc((size_t)BB * 512 * 2);
    __bf16* A3 = (__bf16*)alloc((size_t)BB * 1536 * 2);
    __bf16* A4 = (__bf16*)alloc((size_t)BB * 1024 * 2);
    __bf16* A5 = (__bf16*)alloc((size_t)BB * 1024 * 2);
    float* parts    = (float*)alloc((size_t)BB * 1536 * 4);
    float* logits_p = (float*)alloc((size_t)BB * 128 * 4);
    float* logits_q = (float*)alloc((size_t)BB * 128 * 4);
    float* carry_belief = (float*)alloc((size_t)BB * BEL * 4);
    float* carry_state  = (float*)alloc((size_t)BB * SS * 4);

    float* outp = (float*)d_out;
    size_t o_belief  = 0;
    size_t o_prior_s = (size_t)TT * BB * BEL;
    size_t o_prior_p = o_prior_s + (size_t)TT * BB * SS;
    size_t o_post_s  = o_prior_p + (size_t)TT * BB * SS;
    size_t o_post_p  = o_post_s + (size_t)TT * BB * SS;

    // ---- pack weights (once per launch) ----
    auto packN = [](int Kpad, int N) { return (Kpad >> 5) * (N >> 4) * 32; };
    pack_weights<<<(packN(KEMB, 512) + 255) / 256, 256, 0, stream>>>(W_embed, pWe, SS + AACT, KEMB, 512);
    pack_weights<<<(packN(1024, 1536) + 255) / 256, 256, 0, stream>>>(W_gru, pWg, 1024, 1024, 1536);
    pack_weights<<<(packN(512, 1024) + 255) / 256, 256, 0, stream>>>(W_p1, pWp1, 512, 512, 1024);
    pack_weights<<<(packN(1024, 128) + 255) / 256, 256, 0, stream>>>(W_p2, pWp2, 1024, 1024, 128);
    pack_weights<<<(packN(1536, 1024) + 255) / 256, 256, 0, stream>>>(W_q1, pWq1, 1536, 1536, 1024);
    pack_weights<<<(packN(1024, 128) + 255) / 256, 256, 0, stream>>>(W_q2, pWq2, 1024, 1024, 128);

    hipMemcpyAsync(carry_belief, prev_belief, (size_t)BB * BEL * 4, hipMemcpyDeviceToDevice, stream);
    hipMemcpyAsync(carry_state, prev_state, (size_t)BB * SS * 4, hipMemcpyDeviceToDevice, stream);

    for (int t = 0; t < TT; ++t) {
        const float* act_t = actions + (size_t)t * BB * AACT;
        const float* obs_t = observations + (size_t)t * BB * EE;
        const float* nt_t  = nonterm + (size_t)t * BB;

        build_a0<<<(BB * KEMB + 255) / 256, 256, 0, stream>>>(carry_state, act_t, nt_t, A0);
        cvt_f32_bf16<<<(BB * BEL + 255) / 256, 256, 0, stream>>>(carry_belief, A1, BB, BEL, 1024, 512);

        // hidden = relu(A0 @ W_embed + b) -> A1[:, :512]
        gemm_wmma_bf16<<<gemm_grid(512), 256, 0, stream>>>(
            A0, KEMB, pWe, b_embed, KEMB, 512, ACT_RELU, nullptr, A1, 1024);
        // parts = A1 @ W_gru + b (f32, pre-layernorm)
        gemm_wmma_bf16<<<gemm_grid(1536), 256, 0, stream>>>(
            A1, 1024, pWg, b_gru, 1024, 1536, ACT_NONE, parts, nullptr, 1536);
        // layernorm + GRU -> belief_n (carry, d_out, A2, A3[:, :512])
        gru_ln_kernel<<<BB, 256, 0, stream>>>(
            parts, ln_g, ln_b, carry_belief,
            outp + o_belief + (size_t)t * BB * BEL, A2, A3);
        cvt_f32_bf16<<<(BB * EE + 255) / 256, 256, 0, stream>>>(obs_t, A3, BB, EE, 1536, 512);

        // prior head
        gemm_wmma_bf16<<<gemm_grid(1024), 256, 0, stream>>>(
            A2, 512, pWp1, b_p1, 512, 1024, ACT_ELU, nullptr, A4, 1024);
        gemm_wmma_bf16<<<gemm_grid(128), 256, 0, stream>>>(
            A4, 1024, pWp2, b_p2, 1024, 128, ACT_ELU, logits_p, nullptr, 128);
        softmax_sample<<<BB / 8, 256, 0, stream>>>(
            logits_p, keys_prior + 2 * t,
            outp + o_prior_p + (size_t)t * BB * SS,
            outp + o_prior_s + (size_t)t * BB * SS, nullptr);

        // posterior head
        gemm_wmma_bf16<<<gemm_grid(1024), 256, 0, stream>>>(
            A3, 1536, pWq1, b_q1, 1536, 1024, ACT_RELU, nullptr, A5, 1024);
        gemm_wmma_bf16<<<gemm_grid(128), 256, 0, stream>>>(
            A5, 1024, pWq2, b_q2, 1024, 128, ACT_RELU, logits_q, nullptr, 128);
        softmax_sample<<<BB / 8, 256, 0, stream>>>(
            logits_q, keys_post + 2 * t,
            outp + o_post_p + (size_t)t * BB * SS,
            outp + o_post_s + (size_t)t * BB * SS, carry_state);
    }
}
